// Encoder_31026843746979
// MI455X (gfx1250) — matrix operand order repaired and checked by reading, not compiled, not run on hardware
//
#include <hip/hip_runtime.h>
#include <hip/hip_bf16.h>

// ---------------------------------------------------------------------------
// GraFITi encoder for MI455X (gfx1250, wave32, WMMA bf16 16x16x32).
// All GEMM-like ops (q/k/v/o projections, QK^T scores, P·V, edge MLP) run on
// v_wmma_f32_16x16x32_bf16. Fragment loads are contiguous and unguarded
// (out-of-range rows/cols are clamped; their WMMA outputs are discarded in
// the epilogue, and padded keys multiply against zero probs).
// ---------------------------------------------------------------------------

#define Bc    8
#define Tc    160
#define DIMc  41
#define NKc   128
#define NHc   4
#define NLc   3
#define KTOT  (Tc * DIMc)       // 6560, multiple of 32
#define QMAX  160               // max attention query rows (tc MAB)
#define KP32_BIG  KTOT          // padded key length, big attentions
#define KP32_CAT  64            // padded key length, cat MAB (41 -> 64)

typedef __bf16 bf16_t;
typedef __attribute__((ext_vector_type(16))) __bf16 v16bf_t;
typedef __attribute__((ext_vector_type(8)))  float  v8f_t;

enum {
  EPI_BIAS = 1, EPI_RELU = 2, EPI_RESPRE = 4, EPI_RESPOST = 8,
  EPI_MK = 16, EPI_SCALE = 32, EPI_WF32 = 64, EPI_WBF16 = 128,
  EPI_WTRB = 256   // write bf16 output transposed: C[n][row] (per-head V layout)
};

// ---------------------------------------------------------------------------
// Stable partition: observed entries first (row-major order), equivalent to
// the stable argsort(1-mask). One block per batch; block-wide scans.
// Also writes the tU / tM output segments.
// ---------------------------------------------------------------------------
__global__ __launch_bounds__(256)
void pack_kernel(const float* __restrict__ mask, const float* __restrict__ value,
                 const float* __restrict__ tval, const float* __restrict__ tmask,
                 int* __restrict__ Ti, int* __restrict__ Ci, float* __restrict__ mk,
                 float* __restrict__ Uval, float* __restrict__ Uind,
                 float* __restrict__ outTU, float* __restrict__ outTM)
{
  int b = blockIdx.x;
  int t = threadIdx.x;
  const float* mrow = mask + (long)b * KTOT;
  __shared__ float fred[256];
  __shared__ int   scan[256];
  __shared__ int   bases[2];   // [0]=next observed slot, [1]=next unobserved slot

  float lsum = 0.0f;
  for (int j = t; j < KTOT; j += 256) lsum += (mrow[j] > 0.5f ? 1.0f : 0.0f);
  fred[t] = lsum; __syncthreads();
  for (int s = 128; s > 0; s >>= 1) { if (t < s) fred[t] += fred[t + s]; __syncthreads(); }
  int totalObs = (int)(fred[0] + 0.5f);
  if (t == 0) { bases[0] = 0; bases[1] = totalObs; }
  __syncthreads();

  long bK = (long)b * KTOT;
  for (int s0 = 0; s0 < KTOT; s0 += 256) {
    int j = s0 + t;
    int m = (j < KTOT && mrow[j] > 0.5f) ? 1 : 0;
    scan[t] = m; __syncthreads();
    for (int off = 1; off < 256; off <<= 1) {
      int v = scan[t];
      int o = (t >= off) ? scan[t - off] : 0;
      __syncthreads();
      scan[t] = v + o; __syncthreads();
    }
    int incl = scan[t];
    int excl = incl - m;
    int obsBase = bases[0], unobsBase = bases[1];
    if (j < KTOT) {
      int d = m ? (obsBase + excl) : (unobsBase + (t - excl));
      float mf = (float)m;
      float tv = tval[bK + j] * mf;
      float tm = tmask[bK + j] * mf;
      mk[bK + d]    = mf;
      Ti[bK + d]    = m ? (j / DIMc) : 0;
      Ci[bK + d]    = m ? (j % DIMc) : 0;
      Uval[bK + d]  = value[bK + j] * mf;
      Uind[bK + d]  = 1.0f - mf + tm;
      outTU[bK + d] = tv;
      outTM[bK + d] = tm;
    }
    __syncthreads();
    if (t == 0) {
      int realCount = (KTOT - s0 < 256) ? (KTOT - s0) : 256;
      int tileObs = scan[realCount - 1];
      bases[0] += tileObs;
      bases[1] += realCount - tileObs;
    }
    __syncthreads();
  }
}

// ---------------------------------------------------------------------------
// Weight prep: f32 (Kd x N) row-major -> bf16 transposed (N x Kd) row-major
// so the WMMA B fragment loads two runs of 8 contiguous K values per lane.
// ---------------------------------------------------------------------------
__global__ void w_to_bf16T(const float* __restrict__ W, bf16_t* __restrict__ Wt,
                           int Kd, int N)
{
  int gid = blockIdx.x * blockDim.x + threadIdx.x;
  if (gid >= Kd * N) return;
  int k = gid / N, n = gid - k * N;
  Wt[(long)n * Kd + k] = (bf16_t)W[gid];
}

// ---------------------------------------------------------------------------
// Generic WMMA GEMM: C[m,n] = sum_k A[m, aOff+k] * Bt[n, bOff+k] (+ epilogue).
// One 16x16 C-tile per wave; 8 waves/block share an A tile staged in LDS.
// aOff/bOff step by 32 per head (heads>1 -> per-head attention scores).
// Row/col clamping instead of predicated loads: clamped lanes compute garbage
// that only lands in D rows/cols the epilogue never writes.
// kLen must be a multiple of 32.
// ---------------------------------------------------------------------------
template <int EPI>
__global__ __launch_bounds__(256)
void wmma_gemm(const bf16_t* __restrict__ A, int lda, long aBatch, int aHStep,
               const bf16_t* __restrict__ Bt, int ldb, long bBatch, int bHStep,
               int M, int N, int kLen,
               const float* __restrict__ bias,
               const float* __restrict__ res, int ldr, long resBatch,
               const float* __restrict__ mk, long mkBatch,
               float* __restrict__ Cf, bf16_t* __restrict__ Cb, int ldc, long cBatch,
               float scaleC, int heads)
{
  __shared__ bf16_t As[16 * 32];
  int z = blockIdx.z;
  int b = z / heads, h = z - b * heads;
  int aOff = h * aHStep, bOff = h * bHStep;
  const bf16_t* Ab = A + (long)b * aBatch;
  const bf16_t* Bb = Bt + (long)b * bBatch;
  int m0 = blockIdx.x * 16;
  int wave = threadIdx.x >> 5;
  int n0 = (blockIdx.y * 8 + wave) * 16;
  int lane = threadIdx.x & 31;
  int r = lane & 15;
  bool hi = (lane & 16) != 0;
  int ncol  = n0 + r;
  int ncolc = ncol < N ? ncol : (N - 1);           // clamped: column discarded later
  // staging coordinates: each thread moves one aligned u32 (2 bf16)
  int sIdx = threadIdx.x * 2;
  int sRow = sIdx >> 5, sK = sIdx & 31;
  int sRowC = (m0 + sRow) < M ? (m0 + sRow) : (M - 1);   // clamped row, discarded later
  v8f_t acc = {0.f, 0.f, 0.f, 0.f, 0.f, 0.f, 0.f, 0.f};

  for (int k0 = 0; k0 < kLen; k0 += 32) {
    // cooperative A-tile stage: 256 threads x u32 = 16x32 bf16 tile
    *(unsigned int*)&As[sIdx] =
        *(const unsigned int*)&Ab[(long)sRowC * lda + aOff + k0 + sK];
    __syncthreads();
    if (n0 < N) {  // wave-uniform predicate; EXEC stays all-ones at the WMMA
      v16bf_t af, bfr;
      const bf16_t* ap = &As[r * 32 + (hi ? 8 : 0)];
      #pragma unroll
      for (int e = 0; e < 8; e++) { af[e] = ap[e]; af[e + 8] = ap[16 + e]; }
      const bf16_t* bp = Bb + (long)ncolc * ldb + bOff + k0 + (hi ? 8 : 0);
      #pragma unroll
      for (int e = 0; e < 8; e++) { bfr[e] = bp[e]; bfr[e + 8] = bp[16 + e]; }     // 2x global_load_b128
      __builtin_prefetch((const void*)(bp + 32), 0, 0);  // global_prefetch_b8
      acc = __builtin_amdgcn_wmma_f32_16x16x32_bf16(false, af, false, bfr,
                                                    (short)0, acc, false, false);
    }
    __syncthreads();
  }

  if (n0 >= N || ncol >= N) return;
  const float* resp = (EPI & (EPI_RESPRE | EPI_RESPOST)) ? res + (long)z * resBatch : nullptr;
  const float* mkp  = (EPI & EPI_MK) ? mk + (long)b * mkBatch : nullptr;
  #pragma unroll
  for (int rr = 0; rr < 8; rr++) {
    int row = m0 + (hi ? 8 : 0) + rr;
    if (row >= M) continue;
    float x = acc[rr];
    if (EPI & EPI_SCALE)   x *= scaleC;
    if (EPI & EPI_BIAS)    x += bias[ncol];
    if (EPI & EPI_RESPRE)  x += resp[(long)row * ldr + ncol];
    if (EPI & EPI_RELU)    x = fmaxf(x, 0.0f);
    if (EPI & EPI_RESPOST) x += resp[(long)row * ldr + ncol];
    if (EPI & EPI_MK)      x *= mkp[row];
    long off = (EPI & EPI_WTRB) ? ((long)z * cBatch + (long)ncol * ldc + row)
                                : ((long)z * cBatch + (long)row * ldc + ncol);
    if (EPI & EPI_WF32)               Cf[off] = x;
    if (EPI & (EPI_WBF16 | EPI_WTRB)) Cb[off] = (bf16_t)x;
  }
}

// ---------------------------------------------------------------------------
// Masked, numerically-stable softmax over keys. Mask computed on the fly:
// valid(k) = (midx[k]==q && mk[k]>0)  (midx=Ci for ct, Ti for tc, null=cat).
// Fully-masked rows degenerate to uniform, matching softmax(const -1e10).
// Writes bf16 probs, zero-padded to Kp32 and for padded q rows.
// ---------------------------------------------------------------------------
__global__ __launch_bounds__(256)
void attn_softmax(const float* __restrict__ scores, bf16_t* __restrict__ probs,
                  const int* __restrict__ midx, const float* __restrict__ mk,
                  int Q, int Kk, int Kp32, int Qp)
{
  int q = blockIdx.x, h = blockIdx.y, b = blockIdx.z;
  long rb = ((long)(b * NHc + h) * Qp + q) * Kp32;
  const float* srow = scores + rb;
  bf16_t* prow = probs + rb;
  int t = threadIdx.x;
  if (q >= Q) { for (int k = t; k < Kp32; k += 256) prow[k] = (bf16_t)0.0f; return; }
  __shared__ float red[256];
  const int*   ib = midx ? midx + (long)b * KTOT : nullptr;
  const float* mb = mk + (long)b * KTOT;

  float mx = -3.0e38f;
  for (int k = t; k < Kk; k += 256) {
    float s = srow[k];
    if (midx) { if (!((ib[k] == q) && (mb[k] > 0.5f))) s = -1.0e10f; }
    mx = fmaxf(mx, s);
  }
  red[t] = mx; __syncthreads();
  for (int s2 = 128; s2 > 0; s2 >>= 1) { if (t < s2) red[t] = fmaxf(red[t], red[t + s2]); __syncthreads(); }
  mx = red[0]; __syncthreads();

  float sum = 0.0f;
  for (int k = t; k < Kk; k += 256) {
    float s = srow[k];
    if (midx) { if (!((ib[k] == q) && (mb[k] > 0.5f))) s = -1.0e10f; }
    sum += __expf(s - mx);
  }
  red[t] = sum; __syncthreads();
  for (int s2 = 128; s2 > 0; s2 >>= 1) { if (t < s2) red[t] += red[t + s2]; __syncthreads(); }
  float inv = 1.0f / red[0];

  for (int k = t; k < Kp32; k += 256) {
    float p = 0.0f;
    if (k < Kk) {
      float s = srow[k];
      if (midx) { if (!((ib[k] == q) && (mb[k] > 0.5f))) s = -1.0e10f; }
      p = __expf(s - mx) * inv;
    }
    prow[k] = (bf16_t)p;
  }
}

// ---------------------------------------------------------------------------
// P·V with fused residual: oh[q, h*32+d] = qf[q, h*32+d] + sum_k P[q,k]*V[k,d].
// V comes pre-transposed per head (vpT[b][n][key]) so both fragments load
// two runs of 8 contiguous bf16 per lane, fully unguarded: keys >= Kk multiply against
// zero probs (softmax zero-pads), so stale finite V data contributes nothing.
// ---------------------------------------------------------------------------
__global__ __launch_bounds__(64)
void attn_pv(const bf16_t* __restrict__ probs, int Kp32, int Qp,
             const bf16_t* __restrict__ vpT, long vBatch,
             const float* __restrict__ qf, long qBatch,
             float* __restrict__ ohf, bf16_t* __restrict__ ohb,
             int Q)
{
  int h = blockIdx.y, b = blockIdx.z;
  int m0 = blockIdx.x * 16;
  int wave = threadIdx.x >> 5;
  int n0 = wave * 16;
  int lane = threadIdx.x & 31;
  int r = lane & 15;
  bool hi = (lane & 16) != 0;
  const bf16_t* P = probs + ((long)(b * NHc + h) * Qp) * Kp32;
  int dcol = h * 32 + n0 + r;
  const bf16_t* Vt = vpT + (long)b * vBatch + (long)dcol * KTOT;
  v8f_t acc = {0.f, 0.f, 0.f, 0.f, 0.f, 0.f, 0.f, 0.f};

  for (int k0 = 0; k0 < Kp32; k0 += 32) {
    v16bf_t af, bfr;
    const bf16_t* ap = P + (long)(m0 + r) * Kp32 + k0 + (hi ? 8 : 0);
    #pragma unroll
    for (int e = 0; e < 8; e++) { af[e] = ap[e]; af[e + 8] = ap[16 + e]; }
    const bf16_t* bp = Vt + k0 + (hi ? 8 : 0);
    #pragma unroll
    for (int e = 0; e < 8; e++) { bfr[e] = bp[e]; bfr[e + 8] = bp[16 + e]; }
    acc = __builtin_amdgcn_wmma_f32_16x16x32_bf16(false, af, false, bfr,
                                                  (short)0, acc, false, false);
  }
  #pragma unroll
  for (int rr = 0; rr < 8; rr++) {
    int row = m0 + (hi ? 8 : 0) + rr;
    if (row >= Q) continue;
    long off = (long)b * qBatch + (long)row * NKc + dcol;
    float x = qf[off] + acc[rr];
    ohf[off] = x; ohb[off] = (bf16_t)x;
  }
}

// ---------------------------------------------------------------------------
// Small elementwise / gather kernels.
// ---------------------------------------------------------------------------
__global__ void edge_init_k(const float* __restrict__ Uval, const float* __restrict__ Uind,
                            const float* __restrict__ mk, const float* __restrict__ W,
                            const float* __restrict__ bias,
                            float* __restrict__ Uf, bf16_t* __restrict__ Ub)
{
  long gid = (long)blockIdx.x * blockDim.x + threadIdx.x;
  if (gid >= (long)Bc * KTOT * NKc) return;
  int n = gid & (NKc - 1); long bk = gid >> 7;
  float x = Uval[bk] * W[n] + Uind[bk] * W[NKc + n] + bias[n];
  x = fmaxf(x, 0.0f) * mk[bk];
  Uf[gid] = x; Ub[gid] = (bf16_t)x;
}

__global__ void time_init_k(const float* __restrict__ cx, const float* __restrict__ W,
                            const float* __restrict__ bias, bf16_t* __restrict__ Temb)
{
  long gid = (long)blockIdx.x * blockDim.x + threadIdx.x;
  if (gid >= (long)Bc * Tc * NKc) return;
  int n = gid & (NKc - 1); long bt = gid >> 7;
  Temb[gid] = (bf16_t)sinf(cx[bt] * W[n] + bias[n]);
}

__global__ void chan_init_k(const float* __restrict__ W, const float* __restrict__ bias,
                            bf16_t* __restrict__ Cemb)
{
  long gid = (long)blockIdx.x * blockDim.x + threadIdx.x;
  if (gid >= (long)Bc * DIMc * NKc) return;
  int n = gid & (NKc - 1); long bc = gid >> 7;
  int c = (int)(bc % DIMc);
  Cemb[gid] = (bf16_t)fmaxf(W[c * NKc + n] + bias[n], 0.0f);
}

__global__ void gather_rows(const bf16_t* __restrict__ src, long srcBatch,
                            const int* __restrict__ idx,
                            bf16_t* __restrict__ dst1, int ld1,
                            bf16_t* __restrict__ dst2, int ld2)
{
  long gid = (long)blockIdx.x * blockDim.x + threadIdx.x;
  if (gid >= (long)Bc * KTOT * NKc) return;
  int n = gid & (NKc - 1); long bk = gid >> 7;
  int k = (int)(bk % KTOT), b = (int)(bk / KTOT);
  int row = idx[(long)b * KTOT + k];
  bf16_t v = src[(long)b * srcBatch + (long)row * NKc + n];
  long e = (long)b * KTOT + k;
  dst1[e * ld1 + n] = v;
  if (dst2) dst2[e * ld2 + n] = v;
}

__global__ void copy_u(const bf16_t* __restrict__ Ub,
                       bf16_t* __restrict__ dst1, int ld1,
                       bf16_t* __restrict__ dst2, int ld2)
{
  long gid = (long)blockIdx.x * blockDim.x + threadIdx.x;
  if (gid >= (long)Bc * KTOT * NKc) return;
  int n = gid & (NKc - 1); long bk = gid >> 7;
  bf16_t v = Ub[gid];
  dst1[bk * ld1 + n] = v;
  if (dst2) dst2[bk * ld2 + n] = v;
}

__global__ void output_dot(const bf16_t* __restrict__ cat3, const float* __restrict__ W,
                           const float* __restrict__ bias, float* __restrict__ out)
{
  long gid = (long)blockIdx.x * blockDim.x + threadIdx.x;
  if (gid >= (long)Bc * KTOT) return;
  const bf16_t* rowp = cat3 + gid * (3 * NKc);
  float acc = bias[0];
  #pragma unroll 4
  for (int j = 0; j < 3 * NKc; j++) acc += (float)rowp[j] * W[j];
  out[gid] = acc;
}

// ---------------------------------------------------------------------------
// Host orchestration.
// ---------------------------------------------------------------------------
struct LinP { const float *W, *b; };
struct MabP { LinP q, k, v, o; };

static inline int nblk(long n) { return (int)((n + 255) / 256); }

#define GEMM_LAUNCH(EPI_, A_, lda_, aB_, aH_, Bt_, ldb_, bB_, bH_, M_, N_, kL_, \
                    bias_, res_, ldr_, resB_, mk_, mkB_, Cf_, Cb_, ldc_, cB_, sc_, heads_) \
  do {                                                                          \
    int MT_ = ((M_) + 15) / 16, NT_ = ((N_) + 15) / 16;                         \
    dim3 g_(MT_, (NT_ + 7) / 8, (heads_) * Bc);                                 \
    wmma_gemm<EPI_><<<g_, dim3(256), 0, stream>>>(                              \
        A_, lda_, (long)(aB_), aH_, Bt_, ldb_, (long)(bB_), bH_, M_, N_, kL_,   \
        bias_, res_, ldr_, (long)(resB_), mk_, (long)(mkB_),                    \
        Cf_, Cb_, ldc_, (long)(cB_), sc_, heads_);                              \
  } while (0)

extern "C" void kernel_launch(void* const* d_in, const int* in_sizes, int n_in,
                              void* d_out, int out_size, void* d_ws, size_t ws_size,
                              hipStream_t stream)
{
  (void)in_sizes; (void)n_in; (void)out_size; (void)ws_size;
  constexpr int EPI_QPROJ = EPI_BIAS | EPI_WF32 | EPI_WBF16;
  constexpr int EPI_KV    = EPI_BIAS | EPI_WBF16;
  constexpr int EPI_KVT   = EPI_BIAS | EPI_WTRB;   // v-proj: transposed bf16 out
  constexpr int EPI_O     = EPI_BIAS | EPI_RELU | EPI_RESPOST | EPI_WBF16;
  constexpr int EPI_EDGE  = EPI_BIAS | EPI_RELU | EPI_RESPRE | EPI_MK | EPI_WF32 | EPI_WBF16;
  constexpr int EPI_SC    = EPI_SCALE | EPI_WF32;

  const float* in_cx    = (const float*)d_in[0];
  const float* in_val   = (const float*)d_in[1];
  const float* in_mask  = (const float*)d_in[2];
  const float* in_tval  = (const float*)d_in[3];
  const float* in_tmask = (const float*)d_in[4];

  // params pytree leaves: sorted-key (jax tree) order starting at d_in[5]:
  // cat[0..2]{k,o,q,v}{W,b}, chan_init{W,b}, ct[0..2]{...}, edge_init{W,b},
  // edge_nn[0..2]{W,b}, output{W,b}, tc[0..2]{...}, time_init{W,b}
  auto F = [&](int i) { return (const float*)d_in[i]; };
  auto linAt = [&](int i) { LinP l; l.W = F(i); l.b = F(i + 1); return l; };
  auto mabAt = [&](int i) { MabP m; m.k = linAt(i); m.o = linAt(i + 2);
                            m.q = linAt(i + 4); m.v = linAt(i + 6); return m; };
  MabP catW[NLc], ctW[NLc], tcW[NLc];
  LinP chanI, edgeI, edgeN[NLc], outL, timeI;
  int p = 5;
  for (int i = 0; i < NLc; i++) catW[i] = mabAt(p + i * 8);  p += 24;
  chanI = linAt(p); p += 2;
  for (int i = 0; i < NLc; i++) ctW[i]  = mabAt(p + i * 8);  p += 24;
  edgeI = linAt(p); p += 2;
  for (int i = 0; i < NLc; i++) edgeN[i] = linAt(p + i * 2); p += 6;
  outL = linAt(p); p += 2;
  for (int i = 0; i < NLc; i++) tcW[i]  = mabAt(p + i * 8);  p += 24;
  timeI = linAt(p); p += 2;

  // ---- workspace suballocation (~330 MB) ----
  char* wbase = (char*)d_ws; size_t woff = 0;
  auto alloc = [&](size_t bytes) -> void* {
    void* ptr = wbase + woff; woff = (woff + bytes + 255) & ~(size_t)255; return ptr;
  };
  const long NE = (long)Bc * KTOT;
  int*    Ti    = (int*)alloc(NE * 4);
  int*    Ci    = (int*)alloc(NE * 4);
  float*  mkB   = (float*)alloc(NE * 4);
  float*  Uval  = (float*)alloc(NE * 4);
  float*  Uind  = (float*)alloc(NE * 4);
  float*  Uf    = (float*)alloc(NE * NKc * 4);
  bf16_t* Ub    = (bf16_t*)alloc(NE * NKc * 2);
  bf16_t* TembA = (bf16_t*)alloc((long)Bc * Tc * NKc * 2);
  bf16_t* TembB = (bf16_t*)alloc((long)Bc * Tc * NKc * 2);
  bf16_t* CembA = (bf16_t*)alloc((long)Bc * DIMc * NKc * 2);
  bf16_t* CembB = (bf16_t*)alloc((long)Bc * DIMc * NKc * 2);
  bf16_t* Cnew  = (bf16_t*)alloc((long)Bc * DIMc * NKc * 2);
  bf16_t* kv    = (bf16_t*)alloc(NE * 2 * NKc * 2);
  bf16_t* cat3  = (bf16_t*)alloc(NE * 3 * NKc * 2);
  bf16_t* kp    = (bf16_t*)alloc(NE * NKc * 2);
  bf16_t* vpT   = (bf16_t*)alloc((long)Bc * NKc * KTOT * 2);  // [b][n][key]
  float*  qf    = (float*)alloc((long)Bc * QMAX * NKc * 4);
  bf16_t* qb    = (bf16_t*)alloc((long)Bc * QMAX * NKc * 2);
  float*  ohf   = (float*)alloc((long)Bc * QMAX * NKc * 4);
  bf16_t* ohb   = (bf16_t*)alloc((long)Bc * QMAX * NKc * 2);
  float*  scores = (float*)alloc((long)Bc * NHc * QMAX * KP32_BIG * 4);   // 134 MB
  bf16_t* probs  = (bf16_t*)alloc((long)Bc * NHc * QMAX * KP32_BIG * 2);  // 67 MB

  auto transpose = [&](const float* W, int Kd, int N) -> bf16_t* {
    bf16_t* t = (bf16_t*)alloc((size_t)Kd * N * 2);
    w_to_bf16T<<<nblk((long)Kd * N), 256, 0, stream>>>(W, t, Kd, N);
    return t;
  };
  bf16_t* ctT[NLc][4]; bf16_t* tcT[NLc][4]; bf16_t* catT[NLc][4]; bf16_t* edgeT[NLc];
  for (int i = 0; i < NLc; i++) {
    ctT[i][0] = transpose(ctW[i].q.W, NKc, NKc);
    ctT[i][1] = transpose(ctW[i].k.W, 2 * NKc, NKc);
    ctT[i][2] = transpose(ctW[i].v.W, 2 * NKc, NKc);
    ctT[i][3] = transpose(ctW[i].o.W, NKc, NKc);
    tcT[i][0] = transpose(tcW[i].q.W, NKc, NKc);
    tcT[i][1] = transpose(tcW[i].k.W, 2 * NKc, NKc);
    tcT[i][2] = transpose(tcW[i].v.W, 2 * NKc, NKc);
    tcT[i][3] = transpose(tcW[i].o.W, NKc, NKc);
    catT[i][0] = transpose(catW[i].q.W, NKc, NKc);
    catT[i][1] = transpose(catW[i].k.W, NKc, NKc);
    catT[i][2] = transpose(catW[i].v.W, NKc, NKc);
    catT[i][3] = transpose(catW[i].o.W, NKc, NKc);
    edgeT[i]  = transpose(edgeN[i].W, 3 * NKc, NKc);
  }

  float* outF = (float*)d_out;
  float* outTU = outF + NE;
  float* outTM = outF + 2 * NE;

  // ---- pack + initial embeddings ----
  pack_kernel<<<Bc, 256, 0, stream>>>(in_mask, in_val, in_tval, in_tmask,
                                      Ti, Ci, mkB, Uval, Uind, outTU, outTM);
  edge_init_k<<<nblk(NE * NKc), 256, 0, stream>>>(Uval, Uind, mkB, edgeI.W, edgeI.b, Uf, Ub);
  time_init_k<<<nblk((long)Bc * Tc * NKc), 256, 0, stream>>>(in_cx, timeI.W, timeI.b, TembA);
  chan_init_k<<<nblk((long)Bc * DIMc * NKc), 256, 0, stream>>>(chanI.W, chanI.b, CembA);

  const float SCALE = 0.08838834764831845f;  // 1/sqrt(128)

  auto run_mab = [&](const bf16_t* Qin, long qInB, int Qrows,
                     const bf16_t* KVin, int dk, long kvB, int KVrows,
                     const MabP& mw, bf16_t* const* wT,
                     const int* maskIdx, bf16_t* Oout, long oB, int Kp32) {
    int Qp = ((Qrows + 15) / 16) * 16;
    GEMM_LAUNCH(EPI_QPROJ, Qin, NKc, qInB, 0, wT[0], NKc, 0, 0, Qrows, NKc, NKc,
                mw.q.b, (const float*)nullptr, 0, 0, (const float*)nullptr, 0,
                qf, qb, NKc, (long)QMAX * NKc, 1.0f, 1);
    GEMM_LAUNCH(EPI_KV, KVin, dk, kvB, 0, wT[1], dk, 0, 0, KVrows, NKc, dk,
                mw.k.b, (const float*)nullptr, 0, 0, (const float*)nullptr, 0,
                (float*)nullptr, kp, NKc, (long)KTOT * NKc, 1.0f, 1);
    // v-proj written transposed per head: vpT[b][n][key]
    GEMM_LAUNCH(EPI_KVT, KVin, dk, kvB, 0, wT[2], dk, 0, 0, KVrows, NKc, dk,
                mw.v.b, (const float*)nullptr, 0, 0, (const float*)nullptr, 0,
                (float*)nullptr, vpT, KTOT, (long)NKc * KTOT, 1.0f, 1);
    // per-head QK^T (aOff/bOff = h*32, dh=32 -> single WMMA k-step)
    GEMM_LAUNCH(EPI_SC, qb, NKc, (long)QMAX * NKc, 32, kp, NKc, (long)KTOT * NKc, 32,
                Qrows, KVrows, 32, (const float*)nullptr, (const float*)nullptr, 0, 0,
                (const float*)nullptr, 0, scores, (bf16_t*)nullptr, Kp32,
                (long)Qp * Kp32, SCALE, NHc);
    attn_softmax<<<dim3(Qp, NHc, Bc), dim3(256), 0, stream>>>(
        scores, probs, maskIdx, mkB, Qrows, KVrows, Kp32, Qp);
    attn_pv<<<dim3(Qp / 16, NHc, Bc), dim3(64), 0, stream>>>(
        probs, Kp32, Qp, vpT, (long)NKc * KTOT, qf, (long)QMAX * NKc, ohf, ohb, Qrows);
    // o-layer: out = oh + relu(oh @ Wo + bo)
    GEMM_LAUNCH(EPI_O, ohb, NKc, (long)QMAX * NKc, 0, wT[3], NKc, 0, 0, Qrows, NKc, NKc,
                mw.o.b, ohf, NKc, (long)QMAX * NKc, (const float*)nullptr, 0,
                (float*)nullptr, Oout, NKc, oB, 1.0f, 1);
  };

  bf16_t *TembCur = TembA, *TembNxt = TembB, *CembCur = CembA, *CembNxt = CembB;
  for (int L = 0; L < NLc; L++) {
    // k_t = Temb[Ti] -> kv[:,0:128] and cat3[:,128:256]; U -> kv[:,128:256], cat3[:,0:128]
    gather_rows<<<nblk(NE * NKc), 256, 0, stream>>>(TembCur, (long)Tc * NKc, Ti,
                                                    kv, 2 * NKc, cat3 + NKc, 3 * NKc);
    copy_u<<<nblk(NE * NKc), 256, 0, stream>>>(Ub, kv + NKc, 2 * NKc, cat3, 3 * NKc);
    // ct MAB: Cnew = MAB2(Cemb, [k_t, U], C_mask)
    run_mab(CembCur, (long)DIMc * NKc, DIMc, kv, 2 * NKc, (long)KTOT * 2 * NKc, KTOT,
            ctW[L], ctT[L], Ci, Cnew, (long)DIMc * NKc, KP32_BIG);
    // k_c = Cemb_pre[Ci] -> kv[:,0:128] and cat3[:,256:384]
    gather_rows<<<nblk(NE * NKc), 256, 0, stream>>>(CembCur, (long)DIMc * NKc, Ci,
                                                    kv, 2 * NKc, cat3 + 2 * NKc, 3 * NKc);
    // tc MAB: Tnew = MAB2(Temb, [k_c, U], T_mask)
    run_mab(TembCur, (long)Tc * NKc, Tc, kv, 2 * NKc, (long)KTOT * 2 * NKc, KTOT,
            tcW[L], tcT[L], Ti, TembNxt, (long)Tc * NKc, KP32_BIG);
    // edge MLP: U = relu(U + [U,k_t,k_c] @ We + be) * mk
    GEMM_LAUNCH(EPI_EDGE, cat3, 3 * NKc, (long)KTOT * 3 * NKc, 0, edgeT[L], 3 * NKc, 0, 0,
                KTOT, NKc, 3 * NKc, edgeN[L].b, Uf, NKc, (long)KTOT * NKc,
                mkB, (long)KTOT, Uf, Ub, NKc, (long)KTOT * NKc, 1.0f, 1);
    // cat MAB: Cemb = MAB2(Cnew, Cnew, none)
    run_mab(Cnew, (long)DIMc * NKc, DIMc, Cnew, NKc, (long)DIMc * NKc, DIMc,
            catW[L], catT[L], nullptr, CembNxt, (long)DIMc * NKc, KP32_CAT);
    bf16_t* t1 = TembCur; TembCur = TembNxt; TembNxt = t1;
    bf16_t* t2 = CembCur; CembCur = CembNxt; CembNxt = t2;
  }

  // final [U, k_t, k_c] and output projection (384 -> 1)
  gather_rows<<<nblk(NE * NKc), 256, 0, stream>>>(TembCur, (long)Tc * NKc, Ti,
                                                  cat3 + NKc, 3 * NKc, (bf16_t*)nullptr, 0);
  gather_rows<<<nblk(NE * NKc), 256, 0, stream>>>(CembCur, (long)DIMc * NKc, Ci,
                                                  cat3 + 2 * NKc, 3 * NKc, (bf16_t*)nullptr, 0);
  copy_u<<<nblk(NE * NKc), 256, 0, stream>>>(Ub, cat3, 3 * NKc, (bf16_t*)nullptr, 0);
  output_dot<<<nblk(NE), 256, 0, stream>>>(cat3, outL.W, outL.b, outF);
}
